// TFN_8839042695327
// MI455X (gfx1250) — matrix operand order repaired and checked by reading, not compiled, not run on hardware
//
#include <hip/hip_runtime.h>
#include <cmath>
#include <complex>
#include <cstdint>
#include <cstring>
#include <algorithm>

// ---------------------------------------------------------------------------
// MI455X (gfx1250, wave32) TFN point-cloud classifier forward pass.
// All channel-mixing contractions run on v_wmma_f32_16x16x32_f16 with
// fragments built from ds_load_b128 pairs (LDS tiles stored so every
// fragment is two contiguous 16B runs). The SH-kernel tile is staged into
// LDS by the Tensor Data Mover (tensor_load_to_lds + s_wait_tensorcnt).
// ---------------------------------------------------------------------------

#define BATCH 32
#define GSCALE_ 6.23832462504f   /* ln(2) * NUM_SHELLS^2 */

typedef _Float16 half_t;
typedef __attribute__((ext_vector_type(16))) _Float16 v16h;
typedef __attribute__((ext_vector_type(8)))  _Float16 h8v;
typedef __attribute__((ext_vector_type(8)))  float    v8f;
typedef __attribute__((ext_vector_type(4)))  unsigned int u32x4;
typedef __attribute__((ext_vector_type(4)))  int      i32x4;
typedef __attribute__((ext_vector_type(8)))  int      i32x8;

struct PartD { int lf, lk, qoff, fcol; };
struct ConvDesc {
  int C[4];        // input channels per degree (C[1] includes +1 yzx)
  int colOff[4];   // column offset of each lf block in gathered matrix
  int gcols, gpad; // total / padded-to-16 gathered columns
  int U;           // channel count of y_in buffer
  int fan[4];      // dense fan-in per output degree J
  int np[4];
  PartD parts[4][11];
};
struct ConstBlob { float S[32*16]; float P[16*32]; float Q[4096]; };

// ------------------------------ device helpers ------------------------------

__device__ __forceinline__ v8f wmma16x16x32(v16h a, v16h b, v8f c) {
  return __builtin_amdgcn_wmma_f32_16x16x32_f16(false, a, false, b, (short)0, c,
                                                false, false);
}

// Build a 16-half fragment from two contiguous 16-byte runs (ds_load_b128 x2)
__device__ __forceinline__ v16h frag_ld(const half_t* p0, const half_t* p1) {
  v16h r;
  *reinterpret_cast<h8v*>(&r)       = *reinterpret_cast<const h8v*>(p0);
  *(reinterpret_cast<h8v*>(&r) + 1) = *reinterpret_cast<const h8v*>(p1);
  return r;
}

__device__ __forceinline__ void real_sh_dev(float x, float y, float z, float* sh) {
  float x2 = x*x, y2 = y*y, z2 = z*z;
  sh[0]  = 0.28209479177f;
  sh[1]  = 0.4886025119f*y;  sh[2]  = 0.4886025119f*z;  sh[3]  = 0.4886025119f*x;
  sh[4]  = 1.0925484306f*x*y; sh[5] = 1.0925484306f*y*z;
  sh[6]  = 0.3153915653f*(3.f*z2 - 1.f);
  sh[7]  = 1.0925484306f*x*z; sh[8] = 0.5462742153f*(x2 - y2);
  sh[9]  = 0.5900435899f*y*(3.f*x2 - y2);
  sh[10] = 2.8906114426f*x*y*z;
  sh[11] = 0.4570457995f*y*(5.f*z2 - 1.f);
  sh[12] = 0.3731763326f*z*(5.f*z2 - 3.f);
  sh[13] = 0.4570457995f*x*(5.f*z2 - 1.f);
  sh[14] = 1.4453057213f*z*(x2 - y2);
  sh[15] = 0.5900435899f*x*(x2 - 3.f*y2);
}

__device__ void eig3_sorted(const float* a /*xx,xy,xz,yy,yz,zz*/, float* V) {
  float A[3][3] = {{a[0],a[1],a[2]},{a[1],a[3],a[4]},{a[2],a[4],a[5]}};
  float Q[3][3] = {{1,0,0},{0,1,0},{0,0,1}};
  for (int sweep = 0; sweep < 16; ++sweep) {
    for (int pi = 0; pi < 3; ++pi) {
      int p = (pi == 2) ? 1 : 0;
      int q = (pi == 0) ? 1 : 2;
      float apq = A[p][q];
      if (fabsf(apq) < 1e-12f) continue;
      float theta = (A[q][q] - A[p][p]) / (2.f*apq);
      float t = copysignf(1.f, theta) / (fabsf(theta) + sqrtf(theta*theta + 1.f));
      float c = rsqrtf(t*t + 1.f), s = t*c;
      for (int k = 0; k < 3; ++k) {
        float akp = A[k][p], akq = A[k][q];
        A[k][p] = c*akp - s*akq; A[k][q] = s*akp + c*akq;
      }
      for (int k = 0; k < 3; ++k) {
        float apk = A[p][k], aqk = A[q][k];
        A[p][k] = c*apk - s*aqk; A[q][k] = s*apk + c*aqk;
      }
      for (int k = 0; k < 3; ++k) {
        float qkp = Q[k][p], qkq = Q[k][q];
        Q[k][p] = c*qkp - s*qkq; Q[k][q] = s*qkp + c*qkq;
      }
    }
  }
  float ev[3] = {A[0][0], A[1][1], A[2][2]};
  int ord[3] = {0,1,2};
  if (ev[ord[0]] > ev[ord[1]]) { int t = ord[0]; ord[0] = ord[1]; ord[1] = t; }
  if (ev[ord[1]] > ev[ord[2]]) { int t = ord[1]; ord[1] = ord[2]; ord[2] = t; }
  if (ev[ord[0]] > ev[ord[1]]) { int t = ord[0]; ord[0] = ord[1]; ord[1] = t; }
  for (int i = 0; i < 3; ++i)
    for (int j = 0; j < 3; ++j) V[i*3 + j] = Q[i][ord[j]];
}

// ------------------------------ kernels -------------------------------------

__global__ void cvt_kernel(const float* __restrict__ s, half_t* __restrict__ d, int n) {
  int t = blockIdx.x*blockDim.x + threadIdx.x;
  if (t < n) d[t] = (half_t)s[t];
}

__global__ void zero_kernel(float* p, int n) {
  int t = blockIdx.x*blockDim.x + threadIdx.x;
  if (t < n) p[t] = 0.f;
}

// PCA alignment + kd-tree ordering. One block (1024 threads) per batch item.
__global__ void kdtree_kernel(const float* __restrict__ x, float* __restrict__ out) {
  const int b = blockIdx.x, tid = threadIdx.x;
  __shared__ float px[1024], py[1024], pz[1024], key[1024], red[1024];
  __shared__ float covS[6], Vm[9], meanS[3];
  float vx = x[(b*1024 + tid)*3 + 0];
  float vy = x[(b*1024 + tid)*3 + 1];
  float vz = x[(b*1024 + tid)*3 + 2];
  for (int c = 0; c < 3; ++c) {
    red[tid] = (c == 0) ? vx : (c == 1) ? vy : vz; __syncthreads();
    for (int o = 512; o; o >>= 1) { if (tid < o) red[tid] += red[tid + o]; __syncthreads(); }
    if (tid == 0) meanS[c] = red[0]*(1.f/1024.f);
    __syncthreads();
  }
  vx -= meanS[0]; vy -= meanS[1]; vz -= meanS[2];
  const float pr[6] = {vx*vx, vx*vy, vx*vz, vy*vy, vy*vz, vz*vz};
  for (int c = 0; c < 6; ++c) {
    red[tid] = pr[c]; __syncthreads();
    for (int o = 512; o; o >>= 1) { if (tid < o) red[tid] += red[tid + o]; __syncthreads(); }
    if (tid == 0) covS[c] = red[0]*(1.f/1024.f);
    __syncthreads();
  }
  if (tid == 0) eig3_sorted(covS, Vm);
  __syncthreads();
  px[tid] = vx*Vm[0] + vy*Vm[3] + vz*Vm[6];
  py[tid] = vx*Vm[1] + vy*Vm[4] + vz*Vm[7];
  pz[tid] = vx*Vm[2] + vy*Vm[5] + vz*Vm[8];
  __syncthreads();
  for (int d = 0; d < 10; ++d) {          // segmented bitonic sort per level
    int L = 1024 >> d, axis = d % 3;
    key[tid] = (axis == 0) ? px[tid] : (axis == 1) ? py[tid] : pz[tid];
    __syncthreads();
    for (int k = 2; k <= L; k <<= 1) {
      for (int j = k >> 1; j > 0; j >>= 1) {
        int ixj = tid ^ j;
        if (ixj > tid) {
          bool up = ((tid & k) == 0);
          if ((key[tid] > key[ixj]) == up) {
            float t;
            t = key[tid]; key[tid] = key[ixj]; key[ixj] = t;
            t = px[tid];  px[tid]  = px[ixj];  px[ixj]  = t;
            t = py[tid];  py[tid]  = py[ixj];  py[ixj]  = t;
            t = pz[tid];  pz[tid]  = pz[ixj];  pz[ixj]  = t;
          }
        }
        __syncthreads();
      }
    }
  }
  out[(b*1024 + tid)*3 + 0] = px[tid];
  out[(b*1024 + tid)*3 + 1] = py[tid];
  out[(b*1024 + tid)*3 + 2] = pz[tid];
}

__global__ void pool_kernel(const float* __restrict__ in, float* __restrict__ out,
                            int Nt, int r) {
  int t = blockIdx.x*blockDim.x + threadIdx.x;
  if (t >= BATCH*Nt*3) return;
  int c = t % 3, n = (t/3) % Nt, b = t/(3*Nt);
  float s = 0.f;
  for (int j = 0; j < r; ++j) s += in[((long long)b*Nt*r + n*r + j)*3 + c];
  out[t] = s/(float)r;
}

// nearest-32 selection + SH x Gaussian-shell kernel build, one block / target.
__global__ void group_kernel(const float* __restrict__ src, const float* __restrict__ tgt,
                             int Ns, int Nt, float radius,
                             int* __restrict__ idx_out, half_t* __restrict__ kern_out) {
  const int bv = blockIdx.x, b = bv / Nt;
  const int tid = threadIdx.x, nth = blockDim.x;   // 128
  __shared__ float d2[1024];
  __shared__ float rv[128]; __shared__ int ri[128];
  __shared__ int   sel[32];
  __shared__ float Ys[32*16], gsh[32*3], gsum[3];
  const float tx = tgt[bv*3+0], ty = tgt[bv*3+1], tz = tgt[bv*3+2];
  const float* sp = src + (long long)b*Ns*3;
  for (int i = tid; i < Ns; i += nth) {
    float dx = sp[i*3]-tx, dy = sp[i*3+1]-ty, dz = sp[i*3+2]-tz;
    d2[i] = dx*dx + dy*dy + dz*dz;
  }
  __syncthreads();
  for (int it = 0; it < 32; ++it) {
    float bvv = 3.4e38f; int bi = -1;
    for (int i = tid; i < Ns; i += nth) {
      float v = d2[i];
      if (v < bvv) { bvv = v; bi = i; }
    }
    rv[tid] = bvv; ri[tid] = bi; __syncthreads();
    for (int o = 64; o; o >>= 1) {
      if (tid < o) {
        if (ri[tid+o] >= 0 && (rv[tid+o] < rv[tid] ||
            (rv[tid+o] == rv[tid] && ri[tid+o] < ri[tid]))) {
          rv[tid] = rv[tid+o]; ri[tid] = ri[tid+o];
        }
      }
      __syncthreads();
    }
    if (tid == 0) { sel[it] = ri[0]; d2[ri[0]] = 3.4e38f; }
    __syncthreads();
  }
  if (tid < 32) {
    int si = sel[tid];
    float pxp = (sp[si*3+0]-tx)/radius, pyp = (sp[si*3+1]-ty)/radius,
          pzp = (sp[si*3+2]-tz)/radius;
    float dist = sqrtf(pxp*pxp + pyp*pyp + pzp*pzp);
    float inv = 1.f/fmaxf(dist, 1e-8f);
    float sh[16];
    real_sh_dev(pxp*inv, pyp*inv, pzp*inv, sh);
    for (int k = 0; k < 16; ++k) Ys[tid*16 + k] = sh[k];
    float mask = (dist <= 1.f) ? 1.f : 0.f;
    for (int s = 0; s < 3; ++s) {
      float sc = dist - 0.5f*(float)s;
      gsh[tid*3 + s] = __expf(-GSCALE_*sc*sc)*mask;
    }
    idx_out[bv*32 + tid] = si;
  }
  __syncthreads();
  if (tid < 3) {
    float s = 0.f;
    for (int p = 0; p < 32; ++p) s += gsh[p*3 + tid];
    gsum[tid] = s + 1e-8f;
  }
  __syncthreads();
  for (int t = tid; t < 48*32; t += nth) {      // kern layout [(k*3+s)][p], f16
    int ks = t >> 5, p = t & 31;
    int k = ks/3, s = ks - k*3;
    kern_out[(long long)bv*1536 + t] = (half_t)(Ys[p*16+k]*gsh[p*3+s]/gsum[s]);
  }
}

// Fused sh_conv: TDM-stage kernel tile -> gather -> WMMA aggregation (LDS) ->
// CG recoupling -> dense-layer input F(J) in f16.
__global__ void conv_kernel(const float* __restrict__ y_in,
                            const float* __restrict__ pts_src,
                            const float* __restrict__ pts_tgt,
                            const int* __restrict__ idx,
                            const half_t* __restrict__ kern,
                            half_t* __restrict__ F0, half_t* __restrict__ F1,
                            half_t* __restrict__ F2, half_t* __restrict__ F3,
                            const float* __restrict__ Qbuf,
                            ConvDesc dc, int Nt, int Ns, int isL0) {
  extern __shared__ char smem_dyn[];
  float*  agg = (float*)smem_dyn;                  // [48][gpad] f32
  half_t* GT  = (half_t*)(agg + 48*dc.gpad);       // [gpad][32] f16 (transposed)
  half_t* Kt  = GT + (long long)32*dc.gpad;        // [48][32]  f16
  const int bv = blockIdx.x, b = bv / Nt;
  const int tid = threadIdx.x, nth = blockDim.x;

#if __has_builtin(__builtin_amdgcn_tensor_load_to_lds)
  if (tid < 32) {  // wave 0 issues one TDM transfer of the 48x32 f16 tile
    unsigned long long ga =
        (unsigned long long)(uintptr_t)(kern + (long long)bv*1536);
    u32x4 g0;
    g0[0] = 1u;                                   // count=1 (user D#)
    g0[1] = (unsigned int)(uintptr_t)Kt;          // lds_addr
    g0[2] = (unsigned int)ga;                     // global_addr[31:0]
    g0[3] = (unsigned int)((ga >> 32) & 0x01FFFFFFu) | (2u << 30); // type=2
    i32x8 g1;
    g1[0] = 0x00010000;        // data_size = 2 bytes
    g1[1] = 0x06000000;        // tensor_dim0 = 1536 (low 16 bits at [63:48])
    g1[2] = 0x00010000;        // tensor_dim0 hi = 0, tensor_dim1 = 1
    g1[3] = 0x06000000;        // tile_dim0 = 1536
    g1[4] = 0;                 // tile_dim1/2 unused
    g1[5] = 1536;              // tensor_dim0_stride
    g1[6] = 0; g1[7] = 0;
    i32x4 z4 = {0, 0, 0, 0};
#if defined(__clang_major__) && (__clang_major__ >= 23)
    i32x8 z8 = {0, 0, 0, 0, 0, 0, 0, 0};
    __builtin_amdgcn_tensor_load_to_lds(g0, g1, z4, z4, z8, 0);
#else
    __builtin_amdgcn_tensor_load_to_lds(g0, g1, z4, z4, 0);
#endif
    __builtin_amdgcn_s_wait_tensorcnt(0);
  }
#else
  for (int t = tid; t < 1536; t += nth) Kt[t] = kern[(long long)bv*1536 + t];
#endif

  const int* id = idx + bv*32;
  for (int t = tid; t < 32*dc.gpad; t += nth) {
    int col = t >> 5, p = t & 31;
    half_t val = (half_t)0.f;
    if (col < dc.gcols) {
      int lf = 3;
      while (lf > 0 && col < dc.colOff[lf]) --lf;
      int local = col - dc.colOff[lf];
      int C = dc.C[lf];
      int m = local / C, c = local - m*C;
      int sI = id[p];
      const float* sp = pts_src + ((long long)b*Ns + sI)*3;
      float fv;
      if (isL0) {
        fv = (lf == 0) ? 1.0f : ((m == 0) ? sp[1] : (m == 1) ? sp[2] : sp[0]);
      } else if (lf == 1 && c == C - 1) {
        fv = (m == 0) ? sp[1] : (m == 1) ? sp[2] : sp[0];
      } else {
        fv = y_in[(((long long)b*Ns + sI)*16 + lf*lf + m)*dc.U + c];
      }
      val = (half_t)fv;
    }
    GT[(long long)col*32 + p] = val;
  }
  __syncthreads();
  {
    const int wave = tid >> 5, lane = tid & 31, nw = nth >> 5;
    const int Ml = lane & 15, grp = lane >> 4;
    const int ntN = dc.gpad >> 4;
    for (int t = wave; t < 3*ntN; t += nw) {
      int rt = t / ntN, ct = t - rt*ntN;
      const half_t* ar = Kt + (rt*16 + Ml)*32;
      int kb = grp*8;
      v16h af = frag_ld(ar + kb, ar + 16 + kb);
      int coln = ct*16 + Ml, kb2 = grp*8;
      const half_t* br = GT + (long long)coln*32 + kb2;
      v16h bf = frag_ld(br, br + 16);
      v8f acc = {0.f,0.f,0.f,0.f,0.f,0.f,0.f,0.f};
      acc = wmma16x16x32(af, bf, acc);
      int rbase = grp*8;
#pragma unroll
      for (int r = 0; r < 8; ++r)
        agg[(long long)(rt*16 + rbase + r)*dc.gpad + ct*16 + Ml] = acc[r];
    }
  }
  __syncthreads();
  half_t* Fp[4] = {F0, F1, F2, F3};
  for (int J = 0; J < 4; ++J) {
    int fan = dc.fan[J];
    long long row0 = (long long)bv*(2*J + 1);
    half_t* F = Fp[J];
    for (int pi = 0; pi < dc.np[J]; ++pi) {
      PartD pd = dc.parts[J][pi];
      int C = dc.C[pd.lf], ml = 2*pd.lf + 1, nl = 2*pd.lk + 1;
      int cnt = (2*J + 1)*3*C;
      int co = dc.colOff[pd.lf];
      for (int t = tid; t < cnt; t += nth) {
        int M = t/(3*C); int rem = t - M*3*C; int s = rem/C, c = rem - s*C;
        float accu = 0.f;
        const float* Qp = Qbuf + pd.qoff + M*ml*nl;
        for (int m = 0; m < ml; ++m)
          for (int n = 0; n < nl; ++n)
            accu += Qp[m*nl + n] *
                    agg[(long long)((pd.lk*pd.lk + n)*3 + s)*dc.gpad + co + m*C + c];
        F[(row0 + M)*fan + pd.fcol + s*C + c] = (half_t)accu;
      }
    }
  }
  if (tid < 3) {   // post-conv yzx concat into degree-1 fan
    const float* tp = pts_tgt + (long long)bv*3;
    float fv = (tid == 0) ? tp[1] : (tid == 1) ? tp[2] : tp[0];
    int fan = dc.fan[1];
    F1[((long long)bv*3 + tid)*fan + fan - 1] = (half_t)fv;
  }
}

// f16 WMMA GEMM, 32x64 macro-tile, K-step 64 (4 wmma per staging round).
// out[r][n] -> out[(r/rpg)*gs + (ro + r%rpg)*ldc + n].  rows % 32 == 0.
__global__ void wmma_gemm_kernel(const half_t* __restrict__ A, const half_t* __restrict__ W,
                                 const float* __restrict__ bias, float* __restrict__ out,
                                 int rows, int K, int N, int rpg, int gs, int ro, int ldc) {
  __shared__ __align__(16) half_t As[32*64];
  __shared__ __align__(16) half_t Ws[64*64];       // transposed [n][k]
  const int rt = blockIdx.x;
  const int col0 = blockIdx.y*64;
  const int wave = threadIdx.x >> 5, lane = threadIdx.x & 31;
  const int myN = col0 + wave*16;
  const bool act = (myN < N);
  const int Ml = lane & 15, grp = lane >> 4;
  v8f acc0 = {0.f,0.f,0.f,0.f,0.f,0.f,0.f,0.f};
  v8f acc1 = {0.f,0.f,0.f,0.f,0.f,0.f,0.f,0.f};
  for (int k0 = 0; k0 < K; k0 += 64) {
    for (int t = threadIdx.x; t < 32*64; t += blockDim.x) {
      int r = t >> 6, kk = t & 63, gk = k0 + kk;
      As[t] = (gk < K) ? A[(long long)(rt*32 + r)*K + gk] : (half_t)0.f;
    }
    for (int t = threadIdx.x; t < 64*64; t += blockDim.x) {
      int n = t >> 6, kk = t & 63, gn = col0 + n, gk = k0 + kk;
      Ws[t] = (gn < N && gk < K) ? W[(long long)gk*N + gn] : (half_t)0.f;
    }
    if (k0 + 64 < K) {                              // global_prefetch next chunk
      __builtin_prefetch(&A[(long long)(rt*32 + lane)*K + k0 + 64], 0, 0);
      __builtin_prefetch(&W[(long long)(k0 + 64 + lane)*N + col0], 0, 0);
    }
    __syncthreads();
    if (act) {
      const int kb = grp*8;
      const half_t* wbase = Ws + (wave*16 + Ml)*64 + grp*8;
#pragma unroll
      for (int kk = 0; kk < 64; kk += 32) {
        const half_t* a0 = As + Ml*64 + kk + kb;
        const half_t* a1 = As + (16 + Ml)*64 + kk + kb;
        v16h af0 = frag_ld(a0, a0 + 16);
        v16h af1 = frag_ld(a1, a1 + 16);
        v16h bf  = frag_ld(wbase + kk, wbase + kk + 16);
        acc0 = wmma16x16x32(af0, bf, acc0);
        acc1 = wmma16x16x32(af1, bf, acc1);
      }
    }
    __syncthreads();
  }
  if (act) {
    int gcol = myN + Ml;
    if (gcol < N) {
      float bvl = bias ? bias[gcol] : 0.f;
      int rbase = grp*8;
#pragma unroll
      for (int r = 0; r < 8; ++r) {
        int g0r = rt*32 + rbase + r;
        if (g0r < rows) {
          int g = g0r / rpg, rr = g0r - g*rpg;
          out[(long long)g*gs + (long long)(ro + rr)*ldc + gcol] = acc0[r] + bvl;
        }
        int g1r = rt*32 + 16 + rbase + r;
        if (g1r < rows) {
          int g = g1r / rpg, rr = g1r - g*rpg;
          out[(long long)g*gs + (long long)(ro + rr)*ldc + gcol] = acc1[r] + bvl;
        }
      }
    }
  }
}

__global__ void eval_kernel(const float* __restrict__ Y, const float* __restrict__ Sm,
                            float* __restrict__ e, int bvTot, int C) {
  long long t = (long long)blockIdx.x*blockDim.x + threadIdx.x;
  long long total = (long long)bvTot*32*C;
  if (t >= total) return;
  int c = (int)(t % C);
  long long r = t / C;
  int d = (int)(r & 31);
  long long bv = r >> 5;
  const float* yb = Y + bv*16*C + c;
  float acc = 0.f;
#pragma unroll
  for (int k = 0; k < 16; ++k) acc += Sm[d*16 + k]*yb[(long long)k*C];
  e[t] = acc;
}

__global__ void stats_kernel(const float* __restrict__ e, long long total, int C,
                             float* __restrict__ sums, float* __restrict__ sqs) {
  __shared__ float ps[128], pq[128];
  for (int i = threadIdx.x; i < C; i += blockDim.x) { ps[i] = 0.f; pq[i] = 0.f; }
  __syncthreads();
  for (long long t = (long long)blockIdx.x*blockDim.x + threadIdx.x; t < total;
       t += (long long)gridDim.x*blockDim.x) {
    int c = (int)(t % C);
    float v = e[t];
    atomicAdd(&ps[c], v); atomicAdd(&pq[c], v*v);
  }
  __syncthreads();
  for (int i = threadIdx.x; i < C; i += blockDim.x) {
    atomicAdd(&sums[i], ps[i]); atomicAdd(&sqs[i], pq[i]);
  }
}

__global__ void apply_coeff_kernel(const float* __restrict__ e, const float* __restrict__ sums,
                                   const float* __restrict__ sqs, int bvTot, int C,
                                   const float* __restrict__ Pm, float* __restrict__ ynext) {
  int t = blockIdx.x*blockDim.x + threadIdx.x;
  if (t >= bvTot*C) return;
  int bv = t / C, c = t - bv*C;
  float cnt = (float)bvTot*32.f;
  float m = sums[c]/cnt, var = sqs[c]/cnt - m*m, inv = rsqrtf(var + 1e-3f);
  float z[32];
  for (int d = 0; d < 32; ++d)
    z[d] = fmaxf((e[((long long)bv*32 + d)*C + c] - m)*inv, 0.f);
  for (int k = 0; k < 16; ++k) {
    float acc = 0.f;
    for (int d = 0; d < 32; ++d) acc += Pm[k*32 + d]*z[d];
    ynext[((long long)bv*16 + k)*C + c] = acc;
  }
}

__global__ void apply_relu_kernel(float* __restrict__ e, const float* __restrict__ sums,
                                  const float* __restrict__ sqs, long long total, int C) {
  long long t = (long long)blockIdx.x*blockDim.x + threadIdx.x;
  if (t >= total) return;
  int c = (int)(t % C);
  float cnt = (float)(total / C);
  float m = sums[c]/cnt, var = sqs[c]/cnt - m*m, inv = rsqrtf(var + 1e-3f);
  e[t] = fmaxf((e[t] - m)*inv, 0.f);
}

// max-pool over 16 points, dodeca->coeff projection, per-degree norms -> f16
__global__ void head_kernel(const float* __restrict__ z2, const float* __restrict__ Pm,
                            half_t* __restrict__ nh) {
  int t = blockIdx.x*blockDim.x + threadIdx.x;
  if (t >= BATCH*128) return;
  int b = t/128, c = t - b*128;
  float g[32];
  for (int d = 0; d < 32; ++d) {
    float mx = -3.4e38f;
    for (int v = 0; v < 16; ++v)
      mx = fmaxf(mx, z2[(((long long)b*16 + v)*32 + d)*128 + c]);
    g[d] = mx;
  }
  for (int l = 0; l < 4; ++l) {
    float ss = 0.f;
    for (int k = l*l; k < (l+1)*(l+1); ++k) {
      float cc = 0.f;
      for (int d = 0; d < 32; ++d) cc += Pm[k*32 + d]*g[d];
      ss += cc*cc;
    }
    nh[b*512 + l*128 + c] = (half_t)sqrtf(fmaxf(ss, 1e-8f));
  }
}

__global__ void bn_cols_kernel(const float* __restrict__ in, half_t* __restrict__ out,
                               int R, int C) {
  int c = blockIdx.x*blockDim.x + threadIdx.x;
  if (c >= C) return;
  float s = 0.f, s2 = 0.f;
  for (int r = 0; r < R; ++r) { float v = in[r*C + c]; s += v; s2 += v*v; }
  float m = s/(float)R, var = s2/(float)R - m*m, inv = rsqrtf(var + 1e-3f);
  for (int r = 0; r < R; ++r)
    out[r*C + c] = (half_t)fmaxf((in[r*C + c] - m)*inv, 0.f);
}

__global__ void softmax_kernel(const float* __restrict__ logit, float* __restrict__ out) {
  int b = blockIdx.x, t = threadIdx.x;   // 64 threads
  __shared__ float buf[64];
  float v = (t < 40) ? logit[b*40 + t] : -3.4e38f;
  buf[t] = v; __syncthreads();
  for (int o = 32; o; o >>= 1) { if (t < o) buf[t] = fmaxf(buf[t], buf[t+o]); __syncthreads(); }
  float mx = buf[0]; __syncthreads();
  float ev = (t < 40) ? __expf(v - mx) : 0.f;
  buf[t] = ev; __syncthreads();
  for (int o = 32; o; o >>= 1) { if (t < o) buf[t] += buf[t+o]; __syncthreads(); }
  if (t < 40) out[b*40 + t] = ev/buf[0];
}

// ------------------------------ host constants -------------------------------

typedef std::complex<double> cplx;

static double factd(int n) { double r = 1; for (int i = 2; i <= n; ++i) r *= (double)i; return r; }

static double cgc(int j1, int m1, int j2, int m2, int j3, int m3) {
  if (m1 + m2 != m3 || j3 < std::abs(j1 - j2) || j3 > j1 + j2) return 0.0;
  double pre = std::sqrt((2*j3 + 1.0)*factd(j3+j1-j2)*factd(j3-j1+j2)*factd(j1+j2-j3)
                         / factd(j1+j2+j3+1));
  pre *= std::sqrt(factd(j3+m3)*factd(j3-m3)*factd(j1-m1)*factd(j1+m1)*factd(j2-m2)*factd(j2+m2));
  double s = 0.0;
  int k0 = std::max(0, std::max(j2 - j3 - m1, j1 - j3 + m2));
  int k1 = std::min(j1 + j2 - j3, std::min(j1 - m1, j2 + m2));
  for (int k = k0; k <= k1; ++k) {
    double d = factd(k)*factd(j1+j2-j3-k)*factd(j1-m1-k)*factd(j2+m2-k)
             * factd(j3-j2+m1+k)*factd(j3-j1-m2+k);
    s += ((k & 1) ? -1.0 : 1.0)/d;
  }
  return pre*s;
}

static void umat(int l, cplx U[7][7]) {
  for (int i = 0; i < 7; ++i) for (int j = 0; j < 7; ++j) U[i][j] = cplx(0, 0);
  U[l][l] = cplx(1, 0);
  const double is2 = 1.0/std::sqrt(2.0);
  for (int m = 1; m <= l; ++m) {
    double sg = (m % 2) ? -1.0 : 1.0;
    U[l+m][l+m] = cplx(sg*is2, 0);
    U[l+m][l-m] = cplx(is2, 0);
    U[l-m][l-m] = cplx(0, is2);
    U[l-m][l+m] = cplx(0, -sg*is2);
  }
}

static int real_cg_fill(int l1, int l2, int L, float* out) {
  int d1 = 2*l1 + 1, d2 = 2*l2 + 1, dL = 2*L + 1;
  cplx C[7][7][7];
  for (int M = 0; M < dL; ++M)
    for (int a = 0; a < d1; ++a)
      for (int bq = 0; bq < d2; ++bq)
        C[M][a][bq] = cplx(cgc(l1, a - l1, l2, bq - l2, L, M - L), 0);
  cplx U1[7][7], U2[7][7], UL[7][7];
  umat(l1, U1); umat(l2, U2); umat(L, UL);
  cplx Q[7][7][7];
  double sre = 0, sim = 0;
  for (int P = 0; P < dL; ++P)
    for (int p = 0; p < d1; ++p)
      for (int q = 0; q < d2; ++q) {
        cplx acc(0, 0);
        for (int M = 0; M < dL; ++M)
          for (int a = 0; a < d1; ++a)
            for (int bq = 0; bq < d2; ++bq)
              acc += UL[P][M]*C[M][a][bq]*std::conj(U1[p][a])*std::conj(U2[q][bq]);
        Q[P][p][q] = acc;
        sre += std::fabs(acc.real()); sim += std::fabs(acc.imag());
      }
  bool re = (sre >= sim);
  for (int P = 0; P < dL; ++P)
    for (int p = 0; p < d1; ++p)
      for (int q = 0; q < d2; ++q)
        out[(P*d1 + p)*d2 + q] = (float)(re ? Q[P][p][q].real() : Q[P][p][q].imag());
  return dL*d1*d2;
}

static void real_sh_h(double x, double y, double z, double* sh) {
  double x2 = x*x, y2 = y*y, z2 = z*z;
  sh[0]  = 0.28209479177;
  sh[1]  = 0.4886025119*y;  sh[2]  = 0.4886025119*z;  sh[3]  = 0.4886025119*x;
  sh[4]  = 1.0925484306*x*y; sh[5] = 1.0925484306*y*z;
  sh[6]  = 0.3153915653*(3.0*z2 - 1.0);
  sh[7]  = 1.0925484306*x*z; sh[8] = 0.5462742153*(x2 - y2);
  sh[9]  = 0.5900435899*y*(3.0*x2 - y2);
  sh[10] = 2.8906114426*x*y*z;
  sh[11] = 0.4570457995*y*(5.0*z2 - 1.0);
  sh[12] = 0.3731763326*z*(5.0*z2 - 3.0);
  sh[13] = 0.4570457995*x*(5.0*z2 - 1.0);
  sh[14] = 1.4453057213*z*(x2 - y2);
  sh[15] = 0.5900435899*x*(x2 - 3.0*y2);
}

static void pentakis(double V[32][3]) {
  int n = 0;
  const double PHI = (1.0 + std::sqrt(5.0))/2.0;
  const double pm1[2] = {-1.0, 1.0};
  const double pmP[2] = {-PHI, PHI};
  for (int i = 0; i < 2; ++i)
    for (int j = 0; j < 2; ++j) {
      double a = pm1[i], b2 = pmP[j];
      V[n][0] = 0;  V[n][1] = a;  V[n][2] = b2; ++n;
      V[n][0] = a;  V[n][1] = b2; V[n][2] = 0;  ++n;
      V[n][0] = b2; V[n][1] = 0;  V[n][2] = a;  ++n;
    }
  for (int i = 0; i < 2; ++i)
    for (int j = 0; j < 2; ++j)
      for (int k = 0; k < 2; ++k) {
        V[n][0] = pm1[i]; V[n][1] = pm1[j]; V[n][2] = pm1[k]; ++n;
      }
  const double pmI[2] = {-1.0/PHI, 1.0/PHI};
  for (int i = 0; i < 2; ++i)
    for (int j = 0; j < 2; ++j) {
      double a = pmI[i], b2 = pmP[j];
      V[n][0] = 0;  V[n][1] = a;  V[n][2] = b2; ++n;
      V[n][0] = a;  V[n][1] = b2; V[n][2] = 0;  ++n;
      V[n][0] = b2; V[n][1] = 0;  V[n][2] = a;  ++n;
    }
  for (int i = 0; i < 32; ++i) {
    double nn = std::sqrt(V[i][0]*V[i][0] + V[i][1]*V[i][1] + V[i][2]*V[i][2]);
    V[i][0] /= nn; V[i][1] /= nn; V[i][2] /= nn;
  }
}

static void inv16(double A[16][16], double Inv[16][16]) {
  double M[16][32];
  for (int r = 0; r < 16; ++r) {
    for (int c = 0; c < 16; ++c) { M[r][c] = A[r][c]; M[r][16+c] = (r == c) ? 1.0 : 0.0; }
  }
  for (int c = 0; c < 16; ++c) {
    int piv = c;
    for (int r = c + 1; r < 16; ++r) if (std::fabs(M[r][c]) > std::fabs(M[piv][c])) piv = r;
    if (piv != c) for (int j = 0; j < 32; ++j) std::swap(M[c][j], M[piv][j]);
    double d = M[c][c];
    for (int j = 0; j < 32; ++j) M[c][j] /= d;
    for (int r = 0; r < 16; ++r) {
      if (r == c) continue;
      double f = M[r][c];
      for (int j = 0; j < 32; ++j) M[r][j] -= f*M[c][j];
    }
  }
  for (int r = 0; r < 16; ++r) for (int c = 0; c < 16; ++c) Inv[r][c] = M[r][16+c];
}

static ConstBlob g_cb;
static ConvDesc  g_dc[3];

static void build_all() {
  std::memset(&g_cb, 0, sizeof(g_cb));
  double V[32][3];
  pentakis(V);
  double S[32][16];
  for (int v = 0; v < 32; ++v) {
    double sh[16];
    real_sh_h(V[v][0], V[v][1], V[v][2], sh);
    for (int k = 0; k < 16; ++k) { S[v][k] = sh[k]; g_cb.S[v*16 + k] = (float)sh[k]; }
  }
  double G[16][16], Gi[16][16];
  for (int a = 0; a < 16; ++a)
    for (int b = 0; b < 16; ++b) {
      double s = 0;
      for (int v = 0; v < 32; ++v) s += S[v][a]*S[v][b];
      G[a][b] = s;
    }
  inv16(G, Gi);
  for (int k = 0; k < 16; ++k)
    for (int v = 0; v < 32; ++v) {
      double s = 0;
      for (int a = 0; a < 16; ++a) s += Gi[k][a]*S[v][a];
      g_cb.P[k*32 + v] = (float)s;
    }
  int qoff[4][4][4];
  int cur = 0;
  for (int lf = 0; lf < 4; ++lf)
    for (int lk = 0; lk < 4; ++lk)
      for (int J = 0; J < 4; ++J)
        if (std::abs(lf - lk) <= J && J <= lf + lk) {
          qoff[lf][lk][J] = cur;
          cur += real_cg_fill(lf, lk, J, g_cb.Q + cur);
        }
  const int UNITS_[3] = {32, 64, 128};
  for (int i = 0; i < 3; ++i) {
    ConvDesc& d = g_dc[i];
    std::memset(&d, 0, sizeof(d));
    int nlf = (i == 0) ? 2 : 4;
    if (i == 0) { d.C[0] = 1; d.C[1] = 1; d.U = 1; }
    else {
      int Up = UNITS_[i-1];
      for (int l = 0; l < 4; ++l) d.C[l] = Up;
      d.C[1] = Up + 1; d.U = Up;
    }
    int cur2 = 0;
    for (int l = 0; l < nlf; ++l) { d.colOff[l] = cur2; cur2 += (2*l + 1)*d.C[l]; }
    d.gcols = cur2;
    for (int l = nlf; l < 4; ++l) d.colOff[l] = cur2;
    d.gpad = (cur2 + 15) & ~15;
    for (int J = 0; J < 4; ++J) {
      int fcol = 0, np = 0;
      for (int lf = 0; lf < nlf; ++lf)
        for (int lk = 0; lk < 4; ++lk)
          if (std::abs(lf - lk) <= J && J <= lf + lk) {
            d.parts[J][np].lf = lf; d.parts[J][np].lk = lk;
            d.parts[J][np].qoff = qoff[lf][lk][J];
            d.parts[J][np].fcol = fcol;
            fcol += 3*d.C[lf]; ++np;
          }
      d.np[J] = np;
      d.fan[J] = fcol + ((J == 1) ? 1 : 0);
    }
  }
}

// ------------------------------ launch ---------------------------------------

extern "C" void kernel_launch(void* const* d_in, const int* in_sizes, int n_in,
                              void* d_out, int out_size, void* d_ws, size_t ws_size,
                              hipStream_t stream) {
  (void)in_sizes; (void)n_in; (void)out_size; (void)ws_size;
  build_all();

  char* base = (char*)d_ws;
  size_t cur = 0;
  auto alloc = [&](size_t bytes) -> void* {
    void* p = base + cur;
    cur = (cur + bytes + 255) & ~(size_t)255;
    return p;
  };

  const int UNITS_[3] = {32, 64, 128};
  const int NPTS_[4]  = {1024, 256, 64, 16};
  const float RAD[3]  = {0.2f, 0.4f, 0.8f};

  float* dC = (float*)alloc(sizeof(ConstBlob));
  const float* dS = dC;
  const float* dP = dC + 512;
  const float* dQ = dC + 1024;
  hipMemcpyAsync(dC, &g_cb, sizeof(ConstBlob), hipMemcpyHostToDevice, stream);

  // f16 weight copies
  half_t* wh[15];
  int wsz[15];
  const float* wsrc[15];
  for (int i = 0; i < 3; ++i)
    for (int J = 0; J < 4; ++J) {
      wsz[i*4 + J]  = g_dc[i].fan[J]*UNITS_[i];
      wsrc[i*4 + J] = (const float*)d_in[1 + i*5 + J];
    }
  wsz[12] = 512*512; wsrc[12] = (const float*)d_in[16];
  wsz[13] = 512*256; wsrc[13] = (const float*)d_in[18];
  wsz[14] = 256*40;  wsrc[14] = (const float*)d_in[20];
  for (int w = 0; w < 15; ++w) wh[w] = (half_t*)alloc((size_t)wsz[w]*2);

  float* pts[4];
  for (int k = 0; k < 4; ++k) pts[k] = (float*)alloc((size_t)BATCH*NPTS_[k]*3*4);

  int*    idxb  = (int*)alloc((size_t)BATCH*256*32*4);
  half_t* kernb = (half_t*)alloc((size_t)BATCH*256*1536*2);
  half_t* Fb[4];
  for (int J = 0; J < 4; ++J) {
    size_t mx = 0;
    for (int i = 0; i < 3; ++i)
      mx = std::max(mx, (size_t)BATCH*NPTS_[i+1]*(2*J + 1)*g_dc[i].fan[J]);
    Fb[J] = (half_t*)alloc(mx*2);
  }
  float* Yb   = (float*)alloc((size_t)BATCH*256*16*32*4);
  float* eb   = (float*)alloc((size_t)BATCH*256*32*32*4);
  float* stat = (float*)alloc(256*4);
  float* yA   = (float*)alloc((size_t)BATCH*256*16*32*4);
  float* yB   = (float*)alloc((size_t)BATCH*64*16*64*4);
  half_t* nh  = (half_t*)alloc(32*512*2);
  float* h1   = (float*)alloc(32*512*4);
  half_t* h1h = (half_t*)alloc(32*512*2);
  float* h2   = (float*)alloc(32*256*4);
  half_t* h2h = (half_t*)alloc(32*256*2);
  float* logits = (float*)alloc(32*48*4);

  for (int w = 0; w < 15; ++w)
    cvt_kernel<<<(wsz[w] + 255)/256, 256, 0, stream>>>(wsrc[w], wh[w], wsz[w]);

  kdtree_kernel<<<BATCH, 1024, 0, stream>>>((const float*)d_in[0], pts[0]);
  for (int k = 0; k < 3; ++k) {
    int nt = NPTS_[k + 1];
    pool_kernel<<<(BATCH*nt*3 + 255)/256, 256, 0, stream>>>(pts[k], pts[k+1], nt,
                                                            NPTS_[k]/nt);
  }

  const float* yin = nullptr;
  for (int i = 0; i < 3; ++i) {
    ConvDesc& dc = g_dc[i];
    int Nt = NPTS_[i+1], Ns = NPTS_[i], U = UNITS_[i];
    group_kernel<<<BATCH*Nt, 128, 0, stream>>>(pts[i], pts[i+1], Ns, Nt, RAD[i],
                                               idxb, kernb);
    size_t lds = (size_t)48*dc.gpad*4 + ((size_t)32*dc.gpad + 1536)*2;
    conv_kernel<<<BATCH*Nt, 256, lds, stream>>>(yin, pts[i], pts[i+1], idxb, kernb,
                                                Fb[0], Fb[1], Fb[2], Fb[3], dQ, dc,
                                                Nt, Ns, (i == 0) ? 1 : 0);
    const float* bias = (const float*)d_in[5 + i*5];
    for (int J = 0; J < 4; ++J) {
      int rows = BATCH*Nt*(2*J + 1);
      dim3 g(rows/32, (U + 63)/64);
      wmma_gemm_kernel<<<g, 128, 0, stream>>>(Fb[J], wh[i*4 + J],
                                              (J == 0) ? bias : nullptr, Yb,
                                              rows, dc.fan[J], U,
                                              2*J + 1, 16*U, J*J, U);
    }
    int bvTot = BATCH*Nt;
    long long tot = (long long)bvTot*32*U;
    eval_kernel<<<(int)((tot + 255)/256), 256, 0, stream>>>(Yb, dS, eb, bvTot, U);
    zero_kernel<<<1, 256, 0, stream>>>(stat, 256);
    stats_kernel<<<2048, 256, 0, stream>>>(eb, tot, U, stat, stat + 128);
    if (i < 2) {
      float* ynext = (i == 0) ? yA : yB;
      apply_coeff_kernel<<<(bvTot*U + 255)/256, 256, 0, stream>>>(eb, stat, stat + 128,
                                                                  bvTot, U, dP, ynext);
      yin = ynext;
    } else {
      apply_relu_kernel<<<(int)((tot + 255)/256), 256, 0, stream>>>(eb, stat, stat + 128,
                                                                    tot, U);
    }
  }

  head_kernel<<<(BATCH*128 + 255)/256, 256, 0, stream>>>(eb, dP, nh);
  {
    dim3 g(1, (512 + 63)/64);
    wmma_gemm_kernel<<<g, 128, 0, stream>>>(nh, wh[12], (const float*)d_in[17], h1,
                                            32, 512, 512, 1, 512, 0, 512);
  }
  bn_cols_kernel<<<(512 + 63)/64, 64, 0, stream>>>(h1, h1h, 32, 512);
  {
    dim3 g(1, (256 + 63)/64);
    wmma_gemm_kernel<<<g, 128, 0, stream>>>(h1h, wh[13], (const float*)d_in[19], h2,
                                            32, 512, 256, 1, 256, 0, 256);
  }
  bn_cols_kernel<<<(256 + 63)/64, 64, 0, stream>>>(h2, h2h, 32, 256);
  {
    dim3 g(1, 1);
    wmma_gemm_kernel<<<g, 128, 0, stream>>>(h2h, wh[14], (const float*)d_in[21], logits,
                                            32, 256, 40, 1, 40, 0, 40);
  }
  softmax_kernel<<<BATCH, 64, 0, stream>>>(logits, (float*)d_out);
}
